// MultiheadSelfAttention_85383949844872
// MI455X (gfx1250) — matrix operand order repaired and checked by reading, not compiled, not run on hardware
//
#include <hip/hip_runtime.h>
#include <hip/hip_bf16.h>
#include <stdint.h>

// MI455X (gfx1250, wave32) fused varlen MHA:
//   cvt -> [QKV GEMM (async-LDS staged B) + fused RMSNorm] -> flash attention
//       -> out projection (async-LDS staged B)
// Matmuls: v_wmma_f32_16x16x32_bf16 (f32 accumulate).
// Weight tiles staged via global_load_async_to_lds_b128 (ASYNCcnt), double-buffered.

#define CDIM 1024
#define NHEAD 16
#define HDIM 64
#define EPSV 1.1920929e-07f
#define BSTRIDE 72   // staged-row stride in u16 (64 data + 8 pad -> conflict-free banks)

typedef __attribute__((ext_vector_type(16))) __bf16 bf16x16;
typedef __attribute__((ext_vector_type(8)))  float  f32x8;

struct __align__(16) U4 { unsigned x, y, z, w; };
struct __align__(16) U8 { U4 lo, hi; };

static __device__ __forceinline__ bf16x16 frag_from(U4 lo, U4 hi) {
    U8 t; t.lo = lo; t.hi = hi;
    return __builtin_bit_cast(bf16x16, t);
}

static __device__ __forceinline__ unsigned short f2bf(float f) {
    unsigned u = __builtin_bit_cast(unsigned, f);
    u += 0x7FFFu + ((u >> 16) & 1u);   // round-to-nearest-even
    return (unsigned short)(u >> 16);
}

static __device__ __forceinline__ f32x8 zero8() {
    f32x8 z = {0.f, 0.f, 0.f, 0.f, 0.f, 0.f, 0.f, 0.f};
    return z;
}

// Low 32 bits of a flat shared pointer == LDS byte offset (ISA 10.2 aperture rule).
static __device__ __forceinline__ unsigned lds_lo32(const void* p) {
    return (unsigned)(unsigned long long)p;
}

// A fragment (16x32 MxK). Caller bakes row (M = lane&15) into `base`.
// elems 0..7 -> K = k0 + hf*8 + e ; elems 8..15 -> K = k0 + 16 + hf*8 + (e-8)
static __device__ __forceinline__ bf16x16 load_a32(const unsigned short* base, int k0, int hf) {
    U4 lo = *(const U4*)(base + k0 + hf * 8);
    U4 hi = *(const U4*)(base + k0 + 16 + hf * 8);
    return frag_from(lo, hi);
}

// B fragment (32x16 KxN). Caller bakes column (N = lane&15) row-of-B^T into `base`.
// elems 0..7 -> K = k0 + hf*8 + e ; elems 8..15 -> K = k0 + 16 + hf*8 + (e-8)
static __device__ __forceinline__ bf16x16 load_b32(const unsigned short* base, int k0, int hf) {
    U4 lo = *(const U4*)(base + k0 + hf * 8);
    U4 hi = *(const U4*)(base + k0 + 16 + hf * 8);
    return frag_from(lo, hi);
}

static __device__ __forceinline__ f32x8 wmma_bf16(bf16x16 a, bf16x16 b, f32x8 c) {
    return __builtin_amdgcn_wmma_f32_16x16x32_bf16(false, a, false, b, (short)0, c, false, false);
}

// Async-stage one 64-row x 64-col bf16 weight chunk into LDS (row stride BSTRIDE u16).
// 512 x 16B pieces, 2 per thread of a 256-thread block. Tracked by the wave's ASYNCcnt.
static __device__ __forceinline__ void stage_w_async(const unsigned short* wrow0,
                                                     unsigned ldsbase, int tid) {
    #pragma unroll
    for (int i = 0; i < 2; ++i) {
        const int c = tid * 2 + i;          // 0..511
        const int row = c >> 3;             // 0..63
        const int piece = c & 7;            // 8 x 16B per row
        const unsigned short* g = wrow0 + (size_t)row * CDIM + piece * 8;
        const unsigned l = ldsbase + row * (BSTRIDE * 2) + piece * 16;
        asm volatile("global_load_async_to_lds_b128 %0, %1, off"
                     :: "v"(l), "v"(g) : "memory");
    }
}

static __device__ __forceinline__ void wait_async0() {
    asm volatile("s_wait_asynccnt 0" ::: "memory");
}

// ---------------------------------------------------------------- cvt f32->bf16
__global__ void cvt_bf16_kernel(const float* __restrict__ src,
                                unsigned short* __restrict__ dst, int n) {
    int i = blockIdx.x * blockDim.x + threadIdx.x;
    if (i < n) dst[i] = f2bf(src[i]);
}

// ------------------------------------------- QKV projection + fused RMSNorm(q,k)
// grid: (T/128, NHEAD, 3), block 256 = 8 waves; each wave: 16 rows x 64 cols (one head).
// All 8 waves share the head's 64 W rows -> double-buffered async LDS staging.
__global__ __launch_bounds__(256) void qkv_rms_kernel(
    const unsigned short* __restrict__ xb,
    const unsigned short* __restrict__ wq,
    const unsigned short* __restrict__ wk,
    const unsigned short* __restrict__ wv,
    const float* __restrict__ bq,
    const float* __restrict__ bk,
    const float* __restrict__ bv,
    const float* __restrict__ qn_w,
    const float* __restrict__ kn_w,
    unsigned short* __restrict__ qb,
    unsigned short* __restrict__ kb,
    unsigned short* __restrict__ vt,
    int T)
{
    __shared__ __align__(16) unsigned short bstage[2][64 * BSTRIDE];

    const int tid  = threadIdx.x;
    const int lane = tid & 31;
    const int wave = tid >> 5;
    const int m  = lane & 15;
    const int hf = lane >> 4;
    const int rowbase = (blockIdx.x * 8 + wave) * 16;
    const int head = blockIdx.y;
    const int mat  = blockIdx.z;

    const unsigned short* W    = (mat == 0) ? wq : ((mat == 1) ? wk : wv);
    const float*          bias = (mat == 0) ? bq : ((mat == 1) ? bk : bv);

    const unsigned short* arow  = xb + (size_t)(rowbase + m) * CDIM;
    const unsigned short* wbase = W + (size_t)(head * HDIM) * CDIM;

    stage_w_async(wbase, lds_lo32(&bstage[0][0]), tid);   // prologue: chunk 0

    f32x8 acc[4];
    #pragma unroll
    for (int nt = 0; nt < 4; ++nt) acc[nt] = zero8();

    const int NKC = CDIM / 64;
    for (int kc = 0; kc < NKC; ++kc) {
        const int k0 = kc * 64;
        wait_async0();
        __syncthreads();
        if (kc + 1 < NKC)
            stage_w_async(wbase + (k0 + 64), lds_lo32(&bstage[(kc + 1) & 1][0]), tid);
        const unsigned short* bs = &bstage[kc & 1][0];
        #pragma unroll
        for (int kk = 0; kk < 64; kk += 32) {
            bf16x16 a = load_a32(arow, k0 + kk, hf);
            acc[0] = wmma_bf16(a, load_b32(bs + ( 0 + m) * BSTRIDE, kk, hf), acc[0]);
            acc[1] = wmma_bf16(a, load_b32(bs + (16 + m) * BSTRIDE, kk, hf), acc[1]);
            acc[2] = wmma_bf16(a, load_b32(bs + (32 + m) * BSTRIDE, kk, hf), acc[2]);
            acc[3] = wmma_bf16(a, load_b32(bs + (48 + m) * BSTRIDE, kk, hf), acc[3]);
        }
    }

    #pragma unroll
    for (int nt = 0; nt < 4; ++nt) {
        float bb = bias[head * HDIM + nt * 16 + m];
        #pragma unroll
        for (int v = 0; v < 8; ++v) acc[nt][v] += bb;
    }

    if (mat < 2) {
        // fused rmsnorm over the 64-wide head: vgpr v holds row (v + 8*hf),
        // its 64 cols live in 4 tiles x 16 lanes of this half-group.
        const float* nw = (mat == 0) ? qn_w : kn_w;
        const float w0 = nw[ 0 + m], w1 = nw[16 + m], w2 = nw[32 + m], w3 = nw[48 + m];
        #pragma unroll
        for (int v = 0; v < 8; ++v) {
            float ss = acc[0][v]*acc[0][v] + acc[1][v]*acc[1][v]
                     + acc[2][v]*acc[2][v] + acc[3][v]*acc[3][v];
            ss += __shfl_xor(ss, 1, 32);
            ss += __shfl_xor(ss, 2, 32);
            ss += __shfl_xor(ss, 4, 32);
            ss += __shfl_xor(ss, 8, 32);
            float r = rsqrtf(ss * (1.0f / HDIM) + EPSV);
            acc[0][v] *= r * w0;
            acc[1][v] *= r * w1;
            acc[2][v] *= r * w2;
            acc[3][v] *= r * w3;
        }
        unsigned short* dst = (mat == 0) ? qb : kb;
        #pragma unroll
        for (int nt = 0; nt < 4; ++nt) {
            int col = head * HDIM + nt * 16 + m;
            #pragma unroll
            for (int v = 0; v < 8; ++v)
                dst[(size_t)(rowbase + v + 8 * hf) * CDIM + col] = f2bf(acc[nt][v]);
        }
    } else {
        // V stored transposed per head-dim: vt[d][token] -> contiguous B-frags for P@V
        #pragma unroll
        for (int nt = 0; nt < 4; ++nt) {
            int d = head * HDIM + nt * 16 + m;
            unsigned short t[8];
            #pragma unroll
            for (int v = 0; v < 8; ++v) t[v] = f2bf(acc[nt][v]);
            U4 u;
            u.x = (unsigned)t[0] | ((unsigned)t[1] << 16);
            u.y = (unsigned)t[2] | ((unsigned)t[3] << 16);
            u.z = (unsigned)t[4] | ((unsigned)t[5] << 16);
            u.w = (unsigned)t[6] | ((unsigned)t[7] << 16);
            *(U4*)(vt + (size_t)d * T + rowbase + 8 * hf) = u;
        }
    }
}

// ---------------------------------------------------------- flash attention
static __device__ __forceinline__ f32x8 score_tile(bf16x16 aq0, bf16x16 aq1,
                                                   const unsigned short* kp, int hf) {
    f32x8 s = zero8();
    s = wmma_bf16(aq0, load_b32(kp, 0, hf), s);
    s = wmma_bf16(aq1, load_b32(kp, 32, hf), s);
    return s;
}

// grid: (LMAX/16=64, NHEAD, nseq), block = 1 wave (32 threads)
__global__ __launch_bounds__(32) void attn_kernel(
    const unsigned short* __restrict__ qb,
    const unsigned short* __restrict__ kb,
    const unsigned short* __restrict__ vt,
    const int* __restrict__ cu,
    unsigned short* __restrict__ ob,
    int T)
{
    __shared__ __align__(16) unsigned short pbuf[16 * 32];  // 16 rows x 32 key probs (bf16)

    const int lane = threadIdx.x;
    const int m  = lane & 15;
    const int hf = lane >> 4;
    const int seq  = blockIdx.z;
    const int head = blockIdx.y;
    const int qt   = blockIdx.x;

    const int s0 = cu[seq];
    const int L  = cu[seq + 1] - s0;
    const int qbase = qt * 16;
    if (qbase >= L) return;

    const int qrow = s0 + ((qbase + m < L) ? (qbase + m) : (L - 1));
    const unsigned short* qp = qb + (size_t)qrow * CDIM + head * HDIM;
    const bf16x16 aq0 = load_a32(qp, 0, hf);
    const bf16x16 aq1 = load_a32(qp, 32, hf);

    float Mv[8], Sv[8];
    f32x8 acc[4];
    #pragma unroll
    for (int v = 0; v < 8; ++v) { Mv[v] = -1e30f; Sv[v] = 0.f; }
    #pragma unroll
    for (int nt = 0; nt < 4; ++nt) acc[nt] = zero8();

    const int nkc = (L + 31) >> 5;
    for (int kc = 0; kc < nkc; ++kc) {
        const int kbase = kc * 32;

        // two 16-key score tiles (this wave's column = m)
        const int k0i = kbase + m;
        const int k1i = kbase + 16 + m;
        const unsigned short* kp0 =
            kb + (size_t)(s0 + ((k0i < L) ? k0i : (L - 1))) * CDIM + head * HDIM;
        const unsigned short* kp1 =
            kb + (size_t)(s0 + ((k1i < L) ? k1i : (L - 1))) * CDIM + head * HDIM;
        f32x8 st0 = score_tile(aq0, aq1, kp0, hf);
        f32x8 st1 = score_tile(aq0, aq1, kp1, hf);
        const float mk0 = (k0i < L) ? 0.f : -1e9f;
        const float mk1 = (k1i < L) ? 0.f : -1e9f;
        #pragma unroll
        for (int v = 0; v < 8; ++v) {
            st0[v] = st0[v] * 0.125f + mk0;   // 1/sqrt(64)
            st1[v] = st1[v] * 0.125f + mk1;
        }

        // online softmax per row (row = v + 8*hf), reduce across the 16-lane half-group
        #pragma unroll
        for (int v = 0; v < 8; ++v) {
            float mx = fmaxf(st0[v], st1[v]);
            mx = fmaxf(mx, __shfl_xor(mx, 1, 32));
            mx = fmaxf(mx, __shfl_xor(mx, 2, 32));
            mx = fmaxf(mx, __shfl_xor(mx, 4, 32));
            mx = fmaxf(mx, __shfl_xor(mx, 8, 32));
            const float newM = fmaxf(Mv[v], mx);
            const float corr = __expf(Mv[v] - newM);
            const float p0 = __expf(st0[v] - newM);
            const float p1 = __expf(st1[v] - newM);
            float rs = p0 + p1;
            rs += __shfl_xor(rs, 1, 32);
            rs += __shfl_xor(rs, 2, 32);
            rs += __shfl_xor(rs, 4, 32);
            rs += __shfl_xor(rs, 8, 32);
            Sv[v] = Sv[v] * corr + rs;
            Mv[v] = newM;
            acc[0][v] *= corr;
            acc[1][v] *= corr;
            acc[2][v] *= corr;
            acc[3][v] *= corr;
            const int row = v + 8 * hf;
            pbuf[row * 32 + m]      = f2bf(p0);
            pbuf[row * 32 + 16 + m] = f2bf(p1);
        }
        asm volatile("s_wait_dscnt 0" ::: "memory");

        // re-layout probs (D layout in LDS) -> A fragment for P@V
        const bf16x16 ap = load_a32(pbuf + m * 32, 0, hf);

        #pragma unroll
        for (int nt = 0; nt < 4; ++nt) {
            const unsigned short* vp =
                vt + (size_t)(head * HDIM + nt * 16 + m) * T + s0 + kbase;
            acc[nt] = wmma_bf16(ap, load_b32(vp, 0, hf), acc[nt]);
        }
    }

    #pragma unroll
    for (int v = 0; v < 8; ++v) {
        const int qi = qbase + v + 8 * hf;
        if (qi >= L) continue;
        const float inv = 1.0f / Sv[v];
        const size_t base = (size_t)(s0 + qi) * CDIM + head * HDIM + m;
        ob[base +  0] = f2bf(acc[0][v] * inv);
        ob[base + 16] = f2bf(acc[1][v] * inv);
        ob[base + 32] = f2bf(acc[2][v] * inv);
        ob[base + 48] = f2bf(acc[3][v] * inv);
    }
}

// ---------------------------------------------------------- output projection
// grid: (T/128, 16 col-groups of 64), block 256 = 8 waves; one wave = 16x64 tile.
// Waves share the col-group's 64 Wo rows -> async LDS staging, double-buffered.
__global__ __launch_bounds__(256) void out_proj_kernel(
    const unsigned short* __restrict__ ob,
    const unsigned short* __restrict__ wo,
    const float* __restrict__ bo,
    float* __restrict__ out,
    int T)
{
    __shared__ __align__(16) unsigned short bstage[2][64 * BSTRIDE];

    const int tid  = threadIdx.x;
    const int lane = tid & 31;
    const int wave = tid >> 5;
    const int m  = lane & 15;
    const int hf = lane >> 4;
    const int rowbase = (blockIdx.x * 8 + wave) * 16;
    const int cg = blockIdx.y;

    const unsigned short* arow  = ob + (size_t)(rowbase + m) * CDIM;
    const unsigned short* wbase = wo + (size_t)(cg * 64) * CDIM;

    stage_w_async(wbase, lds_lo32(&bstage[0][0]), tid);

    f32x8 acc[4];
    #pragma unroll
    for (int nt = 0; nt < 4; ++nt) acc[nt] = zero8();

    const int NKC = CDIM / 64;
    for (int kc = 0; kc < NKC; ++kc) {
        const int k0 = kc * 64;
        wait_async0();
        __syncthreads();
        if (kc + 1 < NKC)
            stage_w_async(wbase + (k0 + 64), lds_lo32(&bstage[(kc + 1) & 1][0]), tid);
        const unsigned short* bs = &bstage[kc & 1][0];
        #pragma unroll
        for (int kk = 0; kk < 64; kk += 32) {
            bf16x16 a = load_a32(arow, k0 + kk, hf);
            acc[0] = wmma_bf16(a, load_b32(bs + ( 0 + m) * BSTRIDE, kk, hf), acc[0]);
            acc[1] = wmma_bf16(a, load_b32(bs + (16 + m) * BSTRIDE, kk, hf), acc[1]);
            acc[2] = wmma_bf16(a, load_b32(bs + (32 + m) * BSTRIDE, kk, hf), acc[2]);
            acc[3] = wmma_bf16(a, load_b32(bs + (48 + m) * BSTRIDE, kk, hf), acc[3]);
        }
    }

    #pragma unroll
    for (int nt = 0; nt < 4; ++nt) {
        const int col = cg * 64 + nt * 16 + m;
        const float bb = bo[col];
        #pragma unroll
        for (int v = 0; v < 8; ++v)
            out[(size_t)(rowbase + v + 8 * hf) * CDIM + col] = acc[nt][v] + bb;
    }
}

// -----------------------------------------------------------------------------
extern "C" void kernel_launch(void* const* d_in, const int* in_sizes, int n_in,
                              void* d_out, int out_size, void* d_ws, size_t ws_size,
                              hipStream_t stream)
{
    const float* x    = (const float*)d_in[0];
    const int*   cu   = (const int*)  d_in[1];
    const float* Wq   = (const float*)d_in[2];
    const float* bq   = (const float*)d_in[3];
    const float* Wk   = (const float*)d_in[4];
    const float* bk   = (const float*)d_in[5];
    const float* Wv   = (const float*)d_in[6];
    const float* bv   = (const float*)d_in[7];
    const float* qn_w = (const float*)d_in[8];
    const float* kn_w = (const float*)d_in[9];
    const float* Wo   = (const float*)d_in[10];
    const float* bo   = (const float*)d_in[11];

    const int C = CDIM;
    const int T = in_sizes[0] / C;          // 8192
    const int nseq = in_sizes[1] - 1;       // 16

    char* ws = (char*)d_ws;
    size_t off = 0;
    auto carve = [&](size_t bytes) -> void* {
        void* p = ws + off;
        off = (off + bytes + 255) & ~(size_t)255;
        return p;
    };
    unsigned short* xb  = (unsigned short*)carve((size_t)T * C * 2);
    unsigned short* wqb = (unsigned short*)carve((size_t)C * C * 2);
    unsigned short* wkb = (unsigned short*)carve((size_t)C * C * 2);
    unsigned short* wvb = (unsigned short*)carve((size_t)C * C * 2);
    unsigned short* wob = (unsigned short*)carve((size_t)C * C * 2);
    unsigned short* qb  = (unsigned short*)carve((size_t)T * C * 2);
    unsigned short* kb  = (unsigned short*)carve((size_t)T * C * 2);
    unsigned short* vt  = (unsigned short*)carve((size_t)T * C * 2 + 256); // +pad for tail reads
    unsigned short* ob  = (unsigned short*)carve((size_t)T * C * 2);

    const int nx = T * C, nw = C * C;
    cvt_bf16_kernel<<<(nx + 255) / 256, 256, 0, stream>>>(x,  xb,  nx);
    cvt_bf16_kernel<<<(nw + 255) / 256, 256, 0, stream>>>(Wq, wqb, nw);
    cvt_bf16_kernel<<<(nw + 255) / 256, 256, 0, stream>>>(Wk, wkb, nw);
    cvt_bf16_kernel<<<(nw + 255) / 256, 256, 0, stream>>>(Wv, wvb, nw);
    cvt_bf16_kernel<<<(nw + 255) / 256, 256, 0, stream>>>(Wo, wob, nw);

    qkv_rms_kernel<<<dim3(T / 128, NHEAD, 3), 256, 0, stream>>>(
        xb, wqb, wkb, wvb, bq, bk, bv, qn_w, kn_w, qb, kb, vt, T);

    attn_kernel<<<dim3(64, NHEAD, nseq), 32, 0, stream>>>(qb, kb, vt, cu, ob, T);

    out_proj_kernel<<<dim3(T / 128, NHEAD), 256, 0, stream>>>(
        ob, wob, bo, (float*)d_out, T);
}
